// CASSBlock_35622458753260
// MI455X (gfx1250) — matrix operand order repaired and checked by reading, not compiled, not run on hardware
//
#include <hip/hip_runtime.h>
#include <hip/hip_bf16.h>

// ---- problem constants ----
#define BB   32
#define HH   64
#define WW   64
#define CC   192
#define DIN  384
#define LL   (HH * WW)          // 4096
#define MM   (BB * LL)          // 131072
#define EPSF 1e-5f

typedef __attribute__((ext_vector_type(16))) _Float16 v16h;
typedef __attribute__((ext_vector_type(8)))  _Float16 v8h;
typedef __attribute__((ext_vector_type(8)))  float    v8f;

// -------------------------------------------------------------------------
// WMMA fragment loaders (CDNA5 ISA 7.12.2 layouts, wave32)
// -------------------------------------------------------------------------
// A: 16x32 f16, row-major source, ldk halves per row. Lane half-split:
//   lanes 0-15: K = {0..7, 16..23}; lanes 16-31: K = {8..15, 24..31}
__device__ inline v16h load_frag_a(const _Float16* __restrict__ A, int ldk, int lane) {
    int m  = lane & 15;
    int kb = (lane >> 4) << 3;                 // 0 or 8
    const _Float16* p = A + (long)m * ldk + kb;
    v8h lo = *reinterpret_cast<const v8h*>(p);
    v8h hi = *reinterpret_cast<const v8h*>(p + 16);
    return __builtin_shufflevector(lo, hi, 0,1,2,3,4,5,6,7,8,9,10,11,12,13,14,15);
}

// B: 32x16 f16; weight stored [N,K] K-contiguous. Per lane: fixed n = lane&15,
// K = {0..7, 16..23} (lanes 0-15) or {8..15, 24..31} (lanes 16-31).
__device__ inline v16h load_frag_b(const _Float16* __restrict__ W, int ldk, int lane) {
    int n  = lane & 15;
    int kk = (lane >> 4) << 3;                 // 0 or 8
    const _Float16* p = W + (long)n * ldk + kk;
    v8h lo = *reinterpret_cast<const v8h*>(p);
    v8h hi = *reinterpret_cast<const v8h*>(p + 16);
    return __builtin_shufflevector(lo, hi, 0,1,2,3,4,5,6,7,8,9,10,11,12,13,14,15);
}

// -------------------------------------------------------------------------
// Kernel 1: f32 -> f16 weight conversion (fc1_w and fc2_w, both 384*192)
// -------------------------------------------------------------------------
__global__ __launch_bounds__(256) void cvt_weights_kernel(
    const float* __restrict__ w1, const float* __restrict__ w2,
    _Float16* __restrict__ o1, _Float16* __restrict__ o2, int n) {
    int i = blockIdx.x * 256 + threadIdx.x;
    if (i < n) {
        o1[i] = (_Float16)w1[i];
        o2[i] = (_Float16)w2[i];
    }
}

// -------------------------------------------------------------------------
// Kernel 2: LayerNorm over C, one wave32 per pixel. Emits xn (f16) and
// g[pixel] = mean_c(xn) for the direction selector.
// -------------------------------------------------------------------------
__global__ __launch_bounds__(256) void layernorm_kernel(
    const float* __restrict__ x, const float* __restrict__ nw,
    const float* __restrict__ nb, _Float16* __restrict__ xn,
    float* __restrict__ g) {
    int lane = threadIdx.x & 31;
    long pix = (long)blockIdx.x * 8 + (threadIdx.x >> 5);  // 8 waves/block
    const float* px = x + pix * CC;

    float v[6];
    float s = 0.f;
    #pragma unroll
    for (int j = 0; j < 6; ++j) { v[j] = px[lane + 32 * j]; s += v[j]; }
    #pragma unroll
    for (int o = 16; o > 0; o >>= 1) s += __shfl_xor(s, o, 32);
    float mu = s * (1.f / CC);

    float vs = 0.f;
    #pragma unroll
    for (int j = 0; j < 6; ++j) { float d = v[j] - mu; vs += d * d; }
    #pragma unroll
    for (int o = 16; o > 0; o >>= 1) vs += __shfl_xor(vs, o, 32);
    float rstd = rsqrtf(vs * (1.f / CC) + EPSF);

    _Float16* po = xn + pix * CC;
    float gs = 0.f;
    #pragma unroll
    for (int j = 0; j < 6; ++j) {
        int c = lane + 32 * j;
        float y = (v[j] - mu) * rstd * nw[c] + nb[c];
        po[c] = (_Float16)y;
        gs += y;
    }
    #pragma unroll
    for (int o = 16; o > 0; o >>= 1) gs += __shfl_xor(gs, o, 32);
    if (lane == 0) g[pix] = gs * (1.f / CC);
}

// -------------------------------------------------------------------------
// Kernel 3: direction selector. One block per batch sample.
// -------------------------------------------------------------------------
__device__ inline int refl64(int i) { return i < 0 ? -i : (i > 63 ? 126 - i : i); }

__global__ __launch_bounds__(256) void selector_kernel(
    const float* __restrict__ g,
    const float* __restrict__ w1, const float* __restrict__ b1,
    const float* __restrict__ w2, const float* __restrict__ b2,
    int* __restrict__ dirs) {
    __shared__ float sA[256], sB[256];
    int b = blockIdx.x;
    const float* gb = g + (long)b * LL;

    // grad_h: rows i in [0,65] (padded), cols w in [0,63]
    float sh = 0.f;
    for (int t = threadIdx.x; t < 66 * 64; t += 256) {
        int i = t >> 6, w = t & 63;
        int ri = refl64(i - 1);
        sh += fabsf(gb[ri * 64 + refl64(w + 1)] - gb[ri * 64 + refl64(w - 1)]);
    }
    // grad_v: rows h in [0,63], cols j in [0,65] (padded)
    float sv = 0.f;
    for (int t = threadIdx.x; t < 64 * 66; t += 256) {
        int h = t / 66, j = t % 66;
        int rj = refl64(j - 1);
        sv += fabsf(gb[refl64(h + 1) * 64 + rj] - gb[refl64(h - 1) * 64 + rj]);
    }
    sA[threadIdx.x] = sh;
    sB[threadIdx.x] = sv;
    __syncthreads();
    for (int o = 128; o > 0; o >>= 1) {
        if (threadIdx.x < o) {
            sA[threadIdx.x] += sA[threadIdx.x + o];
            sB[threadIdx.x] += sB[threadIdx.x + o];
        }
        __syncthreads();
    }
    if (threadIdx.x == 0) {
        float mh = sA[0] * (1.f / 4224.f);
        float mv = sB[0] * (1.f / 4224.f);
        float sc[4] = { mh, mv, 0.5f * (mh + mv), fabsf(mh - mv) };
        float hid[16];
        #pragma unroll
        for (int i = 0; i < 16; ++i) {
            float a = b1[i];
            #pragma unroll
            for (int j = 0; j < 4; ++j) a += sc[j] * w1[i * 4 + j];
            hid[i] = a > 0.f ? a : 0.f;
        }
        float best = -1e30f; int idx = 0;
        #pragma unroll
        for (int k = 0; k < 4; ++k) {
            float a = b2[k];
            #pragma unroll
            for (int i = 0; i < 16; ++i) a += hid[i] * w2[k * 16 + i];
            if (a > best) { best = a; idx = k; }
        }
        dirs[b] = idx;
    }
}

// -------------------------------------------------------------------------
// Kernel 4: gather xn into direction-ordered sequence (16B chunks)
// 24 chunks of 8 halves per pixel.
// -------------------------------------------------------------------------
__global__ __launch_bounds__(256) void gather_kernel(
    const _Float16* __restrict__ xn, const int* __restrict__ dirs,
    _Float16* __restrict__ seq) {
    long gid = (long)blockIdx.x * 256 + threadIdx.x;   // B*L*24 total
    long pix = gid / 24;
    int  ch  = (int)(gid % 24);
    int  b   = (int)(pix >> 12);
    int  l   = (int)(pix & 4095);
    int  d   = dirs[b];
    int src;
    if (d == 0)      src = l;                                   // h-scan
    else if (d == 3) src = ((l & 63) << 6) + 63 - (l >> 6);     // anti-diag
    else             src = ((l & 63) << 6) + (l >> 6);          // v / diag
    const v8h* s = reinterpret_cast<const v8h*>(xn + ((long)b * LL + src) * CC) + ch;
    v8h*       o = reinterpret_cast<v8h*>(seq + pix * CC) + ch;
    *o = *s;
}

// -------------------------------------------------------------------------
// WMMA GEMM: C[M,N] = A[M,K] * W[N,K]^T (+bias).
// Block tile 128(M) x 64(N), 8 waves in a 4(M) x 2(N) grid.
// Each wave: 32x32 output = 2x2 accumulators -> 4 WMMAs per 4 fragment
// loads per K-step (2x the WMMA-per-byte of a 1x2 tiling). K unrolled.
// EPI 0: store f16 (+bias)   EPI 1: store f32 out = x + acc + bias
// -------------------------------------------------------------------------
template<int EPI, int K>
__global__ __launch_bounds__(256) void gemm_kernel(
    const _Float16* __restrict__ A, int lda,
    const _Float16* __restrict__ W,
    const float* __restrict__ bias,
    _Float16* __restrict__ Hout, int ldh,
    const float* __restrict__ Xres,
    float* __restrict__ Out, int ldo) {
    int lane = threadIdx.x & 31;
    int wave = threadIdx.x >> 5;
    int wm = wave >> 1;                         // 0..3
    int wn = wave & 1;                          // 0..1
    long m0 = (long)blockIdx.x * 128 + wm * 32; // wave owns rows m0..m0+31
    int  n0 = blockIdx.y * 64 + wn * 32;        // wave owns cols n0..n0+31

    v8f acc00 = {}, acc01 = {}, acc10 = {}, acc11 = {};
    const _Float16* a0p = A + m0 * lda;
    const _Float16* a1p = A + (m0 + 16) * lda;
    const _Float16* w0p = W + (long)n0 * K;
    const _Float16* w1p = W + (long)(n0 + 16) * K;

    #pragma unroll
    for (int k0 = 0; k0 < K; k0 += 32) {
        v16h a0 = load_frag_a(a0p + k0, lda, lane);
        v16h a1 = load_frag_a(a1p + k0, lda, lane);
        v16h b0 = load_frag_b(w0p + k0, K, lane);
        v16h b1 = load_frag_b(w1p + k0, K, lane);
        acc00 = __builtin_amdgcn_wmma_f32_16x16x32_f16(
            false, a0, false, b0, (short)0, acc00, false, false);
        acc01 = __builtin_amdgcn_wmma_f32_16x16x32_f16(
            false, a0, false, b1, (short)0, acc01, false, false);
        acc10 = __builtin_amdgcn_wmma_f32_16x16x32_f16(
            false, a1, false, b0, (short)0, acc10, false, false);
        acc11 = __builtin_amdgcn_wmma_f32_16x16x32_f16(
            false, a1, false, b1, (short)0, acc11, false, false);
    }

    // C/D layout: VGPR r -> M = r + 8*(lane>=16); N = lane&15
    int mrow = (lane >> 4) * 8;
    int ncol = lane & 15;
    int na = n0 + ncol;
    int nb = n0 + 16 + ncol;
    #pragma unroll
    for (int r = 0; r < 8; ++r) {
        long ma = m0 + mrow + r;
        long mb = ma + 16;
        if (EPI == 0) {
            Hout[ma * ldh + na] = (_Float16)(acc00[r] + bias[na]);
            Hout[ma * ldh + nb] = (_Float16)(acc01[r] + bias[nb]);
            Hout[mb * ldh + na] = (_Float16)(acc10[r] + bias[na]);
            Hout[mb * ldh + nb] = (_Float16)(acc11[r] + bias[nb]);
        } else {
            Out[ma * ldo + na] = Xres[ma * ldo + na] + acc00[r] + bias[na];
            Out[ma * ldo + nb] = Xres[ma * ldo + nb] + acc01[r] + bias[nb];
            Out[mb * ldo + na] = Xres[mb * ldo + na] + acc10[r] + bias[na];
            Out[mb * ldo + nb] = Xres[mb * ldo + nb] + acc11[r] + bias[nb];
        }
    }
}

// -------------------------------------------------------------------------
// Kernel 6: depthwise conv3 (zero pad) + bias + exact GELU, f16 in/out.
// One thread = 8 channels at one (b,l). 48 chunks per position.
// -------------------------------------------------------------------------
__global__ __launch_bounds__(256) void conv_gelu_kernel(
    const _Float16* __restrict__ h, const float* __restrict__ cw,
    const float* __restrict__ cb, _Float16* __restrict__ a) {
    long gid = (long)blockIdx.x * 256 + threadIdx.x;   // M*48 total
    long pl = gid / 48;
    int  ch = (int)(gid % 48);
    int  d0 = ch * 8;
    int  l  = (int)(pl & 4095);
    const _Float16* base = h + pl * DIN + d0;

    v8h cur = *reinterpret_cast<const v8h*>(base);
    v8h prv = {};
    v8h nxt = {};
    if (l > 0)    prv = *reinterpret_cast<const v8h*>(base - DIN);
    if (l < 4095) nxt = *reinterpret_cast<const v8h*>(base + DIN);

    v8h out;
    #pragma unroll
    for (int j = 0; j < 8; ++j) {
        int d = d0 + j;
        float vv = cw[d * 3 + 0] * (float)prv[j]
                 + cw[d * 3 + 1] * (float)cur[j]
                 + cw[d * 3 + 2] * (float)nxt[j]
                 + cb[d];
        float ge = 0.5f * vv * (1.f + erff(vv * 0.70710678118f));
        out[j] = (_Float16)ge;
    }
    *reinterpret_cast<v8h*>(a + pl * DIN + d0) = out;
}

// -------------------------------------------------------------------------
// Host launcher
// -------------------------------------------------------------------------
extern "C" void kernel_launch(void* const* d_in, const int* in_sizes, int n_in,
                              void* d_out, int out_size, void* d_ws, size_t ws_size,
                              hipStream_t stream) {
    const float* x      = (const float*)d_in[0];
    const float* norm_w = (const float*)d_in[1];
    const float* norm_b = (const float*)d_in[2];
    const float* sel_w1 = (const float*)d_in[3];
    const float* sel_b1 = (const float*)d_in[4];
    const float* sel_w2 = (const float*)d_in[5];
    const float* sel_b2 = (const float*)d_in[6];
    const float* fc1_w  = (const float*)d_in[7];
    const float* fc1_b  = (const float*)d_in[8];
    const float* conv_w = (const float*)d_in[9];
    const float* conv_b = (const float*)d_in[10];
    const float* fc2_w  = (const float*)d_in[11];
    const float* fc2_b  = (const float*)d_in[12];
    float* out = (float*)d_out;

    // workspace layout (256B aligned)
    char* ws = (char*)d_ws;
    size_t off = 0;
    auto alloc = [&](size_t bytes) {
        char* p = ws + off;
        off += (bytes + 255) & ~(size_t)255;
        return p;
    };
    _Float16* xn    = (_Float16*)alloc((size_t)MM * CC * 2);    // 50 MB
    float*    g     = (float*)   alloc((size_t)MM * 4);         // 0.5 MB
    int*      dirs  = (int*)     alloc(BB * 4);
    _Float16* fc1h  = (_Float16*)alloc((size_t)DIN * CC * 2);
    _Float16* fc2h  = (_Float16*)alloc((size_t)CC * DIN * 2);
    _Float16* seq   = (_Float16*)alloc((size_t)MM * CC * 2);    // 50 MB
    _Float16* hbuf  = (_Float16*)alloc((size_t)MM * DIN * 2);   // 96 MB
    _Float16* abuf  = (_Float16*)alloc((size_t)MM * DIN * 2);   // 96 MB
    (void)ws_size;

    // 1) weight conversion (both weights are DIN*CC elems)
    cvt_weights_kernel<<<(DIN * CC + 255) / 256, 256, 0, stream>>>(
        fc1_w, fc2_w, fc1h, fc2h, DIN * CC);

    // 2) LayerNorm (8 pixels per 256-thread block)
    layernorm_kernel<<<MM / 8, 256, 0, stream>>>(x, norm_w, norm_b, xn, g);

    // 3) direction selector
    selector_kernel<<<BB, 256, 0, stream>>>(g, sel_w1, sel_b1, sel_w2, sel_b2, dirs);

    // 4) gather into direction-ordered sequence
    gather_kernel<<<(long)MM * 24 / 256, 256, 0, stream>>>(xn, dirs, seq);

    // 5) GEMM1: [M,192] x [384,192]^T -> h f16 (+fc1_b)
    {
        dim3 grid(MM / 128, DIN / 64);
        gemm_kernel<0, CC><<<grid, 256, 0, stream>>>(
            seq, CC, fc1h, fc1_b, hbuf, DIN, nullptr, nullptr, 0);
    }

    // 6) depthwise conv3 + GELU
    conv_gelu_kernel<<<(long)MM * 48 / 256, 256, 0, stream>>>(
        hbuf, conv_w, conv_b, abuf);

    // 7) GEMM2: [M,384] x [192,384]^T + fc2_b + residual -> out f32
    {
        dim3 grid(MM / 128, CC / 64);
        gemm_kernel<1, DIN><<<grid, 256, 0, stream>>>(
            abuf, DIN, fc2h, fc2_b, nullptr, 0, x, out, CC);
    }
}
